// PairRepresentation_56014963475098
// MI455X (gfx1250) — matrix operand order repaired and checked by reading, not compiled, not run on hardware
//
#include <hip/hip_runtime.h>
#include <math.h>

// ---------------------------------------------------------------------------
// Problem constants (from reference): B=2, SQ=2048, SK=512, HID=1024, NH=16, HD=64
// ---------------------------------------------------------------------------
#define B_    2
#define SQ_   2048
#define SK_   512
#define HID_  1024
#define NH_   16
#define HD_   64

typedef __bf16 bf16_t;
typedef __attribute__((ext_vector_type(16))) __bf16 v16bf;
typedef __attribute__((ext_vector_type(8)))  __bf16 v8bf;
typedef __attribute__((ext_vector_type(8)))  float  v8f;

// round-to-nearest-even f32 -> bf16
static __device__ __forceinline__ bf16_t f2bf(float f) {
    unsigned u = __builtin_bit_cast(unsigned, f);
    unsigned r = (u + 0x7FFFu + ((u >> 16) & 1u)) >> 16;
    return __builtin_bit_cast(bf16_t, (unsigned short)r);
}
static __device__ __forceinline__ unsigned short f2bf_bits(float f) {
    unsigned u = __builtin_bit_cast(unsigned, f);
    return (unsigned short)((u + 0x7FFFu + ((u >> 16) & 1u)) >> 16);
}

// LDS byte offset of a generic shared pointer (flat LDS addr = addr[31:0])
static __device__ __forceinline__ unsigned lds_off(const void* p) {
    return (unsigned)(unsigned long long)p;
}

// async global->LDS 16-byte copy, tracked with ASYNCcnt
static __device__ __forceinline__ void async_copy_b128(unsigned ldsByteOff,
                                                       const void* gaddr) {
    asm volatile("global_load_async_to_lds_b128 %0, %1, off"
                 :: "v"(ldsByteOff), "v"((unsigned long long)gaddr)
                 : "memory");
}
static __device__ __forceinline__ void async_wait0() {
    asm volatile("s_wait_asynccnt 0" ::: "memory");
}

// ---------------------------------------------------------------------------
// Kernel 1: f32 -> bf16 cast, 4 elements / thread
// ---------------------------------------------------------------------------
__global__ void xattn_cast_bf16(const float4* __restrict__ in,
                                ushort4* __restrict__ out, int n4) {
    int i = blockIdx.x * blockDim.x + threadIdx.x;
    if (i < n4) {
        float4 f = in[i];
        ushort4 o;
        o.x = f2bf_bits(f.x); o.y = f2bf_bits(f.y);
        o.z = f2bf_bits(f.z); o.w = f2bf_bits(f.w);
        out[i] = o;
    }
}

// ---------------------------------------------------------------------------
// Kernel 2: D[m,n] = sum_k A[m,k] * W[n,k]   (nn.Linear, bias-free)
// Block = 8 waves sharing one 16-row A tile, async-staged to LDS (32*K bytes,
// one flat contiguous copy). Each wave computes one 16x16 N-tile via
// v_wmma_f32_16x16x32_bf16; B-operand streams from global (contiguous in K).
// transposeV==1 stores D transposed into Vt[B][HID][SK].
// ---------------------------------------------------------------------------
__global__ __launch_bounds__(256)
void xattn_gemm_bt(const bf16_t* __restrict__ A, const bf16_t* __restrict__ W,
                   bf16_t* __restrict__ out, int Ntile, int K, int transposeV) {
    extern __shared__ char smem_g[];
    bf16_t* sA = (bf16_t*)smem_g;                 // [16][K]

    const int tid  = threadIdx.x;
    const int lane = tid & 31;
    const int wid  = tid >> 5;
    const int blocksPerRow = Ntile >> 3;          // 8 N-tiles per block
    const int mt = blockIdx.x / blocksPerRow;
    const int m0 = mt * 16;
    const int n0 = ((blockIdx.x % blocksPerRow) * 8 + wid) * 16;
    const int hl  = lane >> 4;                    // ISA A/B lane-half selector
    const int l16 = lane & 15;

    // ---- async stage A[m0..m0+15][0..K) -> LDS (contiguous 32*K bytes) ----
    {
        const int chunks = (16 * K) >> 3;         // b128 chunks (8 bf16 each)
        const bf16_t* gA = A + (size_t)m0 * K;
        for (int c = tid; c < chunks; c += 256)
            async_copy_b128(lds_off(sA + c * 8), gA + c * 8);
        async_wait0();
    }
    __syncthreads();

    // ---- WMMA mainloop: A from LDS, B from global ----
    const bf16_t* bRow = W + (size_t)(n0 + l16) * K;   // B column n = W row n
    const bf16_t* aLds = sA + (size_t)l16 * K;

    v8f acc = {};
    for (int kb = 0; kb < K; kb += 32) {
        v16bf a, b;
        // A 16x32 bf16 layout: lanes 0-15 K=[0..8)+[16..24), lanes 16-31 K=[8..16)+[24..32)
        ((v8bf*)&a)[0] = *(const v8bf*)(aLds + kb + hl * 8);
        ((v8bf*)&a)[1] = *(const v8bf*)(aLds + kb + hl * 8 + 16);
        // B 32x16 bf16 layout: same K split as A (lanes 0-15 K=[0..8)+[16..24), lanes 16-31 K=[8..16)+[24..32))
        ((v8bf*)&b)[0] = *(const v8bf*)(bRow + kb + hl * 8);
        ((v8bf*)&b)[1] = *(const v8bf*)(bRow + kb + hl * 8 + 16);
        __builtin_prefetch(bRow + kb + 64, 0, 3);      // near-scope speculative
        acc = __builtin_amdgcn_wmma_f32_16x16x32_bf16(
                  false, a, false, b, (short)0, acc, false, false);
    }

    // C/D layout: VGPR r -> M = r + 8*hl, N = l16
    if (!transposeV) {
        const int N = Ntile * 16;
        for (int r = 0; r < 8; ++r) {
            int m = m0 + hl * 8 + r;
            int n = n0 + l16;
            out[(size_t)m * N + n] = f2bf(acc[r]);
        }
    } else {
        for (int r = 0; r < 8; ++r) {
            int m = m0 + hl * 8 + r;      // m indexes B*SK (keys)
            int n = n0 + l16;             // n indexes HID
            int b    = m >> 9;            // / SK_
            int kkey = m & (SK_ - 1);
            out[((size_t)b * HID_ + n) * SK_ + kkey] = f2bf(acc[r]);
        }
    }
}

// ---------------------------------------------------------------------------
// Kernel 3: scores[b,h,q,k] = (Q_head[q,:] . K_head[k,:]) / 8
// Per (b,h): M=2048, N=512, K=64 (2 WMMA k-steps). f32 output, pre-scaled.
// ---------------------------------------------------------------------------
__global__ __launch_bounds__(256)
void xattn_scores(const bf16_t* __restrict__ Q, const bf16_t* __restrict__ Kb,
                  float* __restrict__ scores) {
    const int lane = threadIdx.x & 31;
    const int wid  = threadIdx.x >> 5;
    const int tile = blockIdx.x * 8 + wid;          // 2^17 tiles total
    const int kt = tile & 31;                       // SK/16 = 32
    const int qt = (tile >> 5) & 127;               // SQ/16 = 128
    const int h  = (tile >> 12) & 15;
    const int b  = tile >> 16;
    const int hl = lane >> 4, l16 = lane & 15;

    const bf16_t* aRow = Q  + (size_t)(b * SQ_ + qt * 16 + l16) * HID_ + h * HD_;
    const bf16_t* bRow = Kb + (size_t)(b * SK_ + kt * 16 + l16) * HID_ + h * HD_;

    v8f acc = {};
    for (int kb = 0; kb < HD_; kb += 32) {
        v16bf a, bb;
        ((v8bf*)&a)[0] = *(const v8bf*)(aRow + kb + hl * 8);
        ((v8bf*)&a)[1] = *(const v8bf*)(aRow + kb + hl * 8 + 16);
        ((v8bf*)&bb)[0] = *(const v8bf*)(bRow + kb + hl * 8);
        ((v8bf*)&bb)[1] = *(const v8bf*)(bRow + kb + hl * 8 + 16);
        acc = __builtin_amdgcn_wmma_f32_16x16x32_bf16(
                  false, a, false, bb, (short)0, acc, false, false);
    }

    float* o = scores + (((size_t)(b * NH_ + h) * SQ_ + qt * 16) * SK_) + kt * 16;
    for (int r = 0; r < 8; ++r) {
        int m = hl * 8 + r;
        o[(size_t)m * SK_ + l16] = acc[r] * 0.125f;   // 1/sqrt(64)
    }
}

// ---------------------------------------------------------------------------
// Kernel 4: fused softmax + P@V.
// One block (4 waves) per (b, h, 16-query tile):
//   - async-stage 16x512 f32 scores into LDS (contiguous 32 KB)
//   - wave-parallel row max / exp-sum (wave32 __shfl_xor reductions)
//   - write P as bf16 to LDS
//   - 4 waves x (16x16 tile) WMMA over K=512: A from LDS, B from Vt (contig)
// Dynamic LDS: 32 KB scores + 16 KB probs = 48 KB.
// ---------------------------------------------------------------------------
__global__ __launch_bounds__(128)
void xattn_softmax_pv(const float* __restrict__ scores,
                      const int* __restrict__ mask,
                      const bf16_t* __restrict__ Vt,
                      float* __restrict__ ctx) {
    extern __shared__ char smem_s[];
    float*  sS = (float*)smem_s;                       // [16][512] f32
    bf16_t* sP = (bf16_t*)(smem_s + 16 * SK_ * 4);     // [16][512] bf16

    const int tid  = threadIdx.x;
    const int lane = tid & 31;
    const int wid  = tid >> 5;
    const int bid  = blockIdx.x;                       // B*NH*(SQ/16) = 4096
    const int qt = bid & 127;
    const int h  = (bid >> 7) & 15;
    const int b  = bid >> 11;

    const float* sBase = scores + (((size_t)(b * NH_ + h) * SQ_ + qt * 16) * SK_);

    // phase 1: async copy 16x512 f32 strip (contiguous) into LDS
    for (int c = tid; c < (16 * SK_) / 4; c += 128)    // b128 = 4 floats
        async_copy_b128(lds_off(sS + c * 4), sBase + c * 4);
    async_wait0();
    __syncthreads();

    // phase 2: softmax; wave `wid` owns rows wid*4 .. wid*4+3
    const int* mrow = mask + b * SK_;
    for (int rr = 0; rr < 4; ++rr) {
        const int r = wid * 4 + rr;
        float v[16];
        float mx = -3.402823466e38f;
        for (int j = 0; j < 16; ++j) {
            const int col = j * 32 + lane;
            const float mz = (float)mrow[col];
            const float s = sS[r * SK_ + col] + (1.0f - mz) * (-3.402823466e38f);
            v[j] = s;
            mx = fmaxf(mx, s);
        }
        for (int off = 16; off >= 1; off >>= 1)
            mx = fmaxf(mx, __shfl_xor(mx, off, 32));
        float sum = 0.f;
        for (int j = 0; j < 16; ++j) { v[j] = __expf(v[j] - mx); sum += v[j]; }
        for (int off = 16; off >= 1; off >>= 1)
            sum += __shfl_xor(sum, off, 32);
        const float inv = 1.0f / sum;
        for (int j = 0; j < 16; ++j) {
            const int col = j * 32 + lane;
            sP[r * SK_ + col] = f2bf(v[j] * inv);
        }
    }
    __syncthreads();

    // phase 3: 16x64 = 4 waves x 16x16 WMMA tiles, K=512
    const int hl = lane >> 4, l16 = lane & 15;
    const int d0 = wid * 16;
    const bf16_t* vBase = Vt + ((size_t)b * HID_ + h * HD_ + d0 + l16) * SK_;

    v8f acc = {};
    for (int kb = 0; kb < SK_; kb += 32) {
        v16bf a, bb;
        ((v8bf*)&a)[0] = *(const v8bf*)(&sP[l16 * SK_ + kb + hl * 8]);
        ((v8bf*)&a)[1] = *(const v8bf*)(&sP[l16 * SK_ + kb + hl * 8 + 16]);
        ((v8bf*)&bb)[0] = *(const v8bf*)(vBase + kb + hl * 8);
        ((v8bf*)&bb)[1] = *(const v8bf*)(vBase + kb + hl * 8 + 16);
        __builtin_prefetch(vBase + kb + 64, 0, 3);
        acc = __builtin_amdgcn_wmma_f32_16x16x32_bf16(
                  false, a, false, bb, (short)0, acc, false, false);
    }

    float* o = ctx + ((size_t)b * SQ_ + qt * 16) * HID_ + h * HD_ + d0;
    for (int r = 0; r < 8; ++r) {
        int m = hl * 8 + r;
        o[(size_t)m * HID_ + l16] = acc[r];
    }
}

// ---------------------------------------------------------------------------
// Host launcher
// ---------------------------------------------------------------------------
extern "C" void kernel_launch(void* const* d_in, const int* in_sizes, int n_in,
                              void* d_out, int out_size, void* d_ws, size_t ws_size,
                              hipStream_t stream) {
    const float* X    = (const float*)d_in[0];   // [B,SQ,HID]
    const float* E    = (const float*)d_in[1];   // [B,SK,HID]
    const int*   mask = (const int*)  d_in[2];   // [B,SK]
    const float* Wq   = (const float*)d_in[3];   // [HID,HID]
    const float* Wk   = (const float*)d_in[4];
    const float* Wv   = (const float*)d_in[5];

    // outputs: context first, then scores (tuple flattened in return order)
    const size_t CTX_N = (size_t)B_ * SQ_ * HID_;            // 4,194,304
    float* ctxOut    = (float*)d_out;
    float* scoresOut = (float*)d_out + CTX_N;

    // workspace layout (bf16 elements), ~29.4 MB total
    bf16_t* ws  = (bf16_t*)d_ws;
    const size_t XN = (size_t)B_ * SQ_ * HID_;   // 4,194,304
    const size_t EN = (size_t)B_ * SK_ * HID_;   // 1,048,576
    const size_t WN = (size_t)HID_ * HID_;       // 1,048,576
    bf16_t* Xb  = ws;
    bf16_t* Eb  = Xb  + XN;
    bf16_t* Wqb = Eb  + EN;
    bf16_t* Wkb = Wqb + WN;
    bf16_t* Wvb = Wkb + WN;
    bf16_t* Qb  = Wvb + WN;                      // [B,SQ,HID]
    bf16_t* Kb  = Qb  + XN;                      // [B,SK,HID]
    bf16_t* Vt  = Kb  + EN;                      // [B,HID,SK] (transposed)

    // 1) casts (4 elems / thread)
    xattn_cast_bf16<<<(int)(XN / 1024), 256, 0, stream>>>((const float4*)X,  (ushort4*)Xb,  (int)(XN / 4));
    xattn_cast_bf16<<<(int)(EN / 1024), 256, 0, stream>>>((const float4*)E,  (ushort4*)Eb,  (int)(EN / 4));
    xattn_cast_bf16<<<(int)(WN / 1024), 256, 0, stream>>>((const float4*)Wq, (ushort4*)Wqb, (int)(WN / 4));
    xattn_cast_bf16<<<(int)(WN / 1024), 256, 0, stream>>>((const float4*)Wk, (ushort4*)Wkb, (int)(WN / 4));
    xattn_cast_bf16<<<(int)(WN / 1024), 256, 0, stream>>>((const float4*)Wv, (ushort4*)Wvb, (int)(WN / 4));

    // 2) projections; dynamic LDS = 16 rows * K * 2 bytes = 32 KB
    const int ldsGemm = 16 * HID_ * 2;
    // grid = Mt * (Ntile/8): Q -> 256*8, K/V -> 64*8
    xattn_gemm_bt<<<256 * 8, 256, ldsGemm, stream>>>(Xb, Wqb, Qb, HID_ / 16, HID_, 0);
    xattn_gemm_bt<<< 64 * 8, 256, ldsGemm, stream>>>(Eb, Wkb, Kb, HID_ / 16, HID_, 0);
    xattn_gemm_bt<<< 64 * 8, 256, ldsGemm, stream>>>(Eb, Wvb, Vt, HID_ / 16, HID_, 1);

    // 3) scores: B*NH*(SQ/16)*(SK/16) = 131072 wave-tiles / 8 per block
    xattn_scores<<<131072 / 8, 256, 0, stream>>>(Qb, Kb, scoresOut);

    // 4) fused softmax + PV: 4096 blocks, 48 KB dynamic LDS
    const int ldsSm = 16 * SK_ * 4 + 16 * SK_ * 2;
    xattn_softmax_pv<<<B_ * NH_ * (SQ_ / 16), 128, ldsSm, stream>>>(scoresOut, mask, Vt, ctxOut);
}
